// SGLSTMLayer_54357106098566
// MI455X (gfx1250) — matrix operand order repaired and checked by reading, not compiled, not run on hardware
//
#include <hip/hip_runtime.h>
#include <hip/hip_bf16.h>
#include <cstddef>
#include <cstdint>

// ---------------------------------------------------------------------------
// Problem constants (from reference): B=64, T=512, D=256, H=256, L=3
// ---------------------------------------------------------------------------
#define BSZ   64
#define TLEN  512
#define DDIM  256
#define HDIM  256
#define G4    1024            // 4*H
#define MROWS (BSZ * TLEN)    // 32768 rows of x
#define HPAD  264             // padded LDS row stride (ushorts) -> bank-conflict free

typedef __attribute__((ext_vector_type(16))) __bf16 v16bf;
typedef __attribute__((ext_vector_type(8)))  float  v8f;
typedef __attribute__((ext_vector_type(4)))  unsigned int uint4v;

union ABReg { v16bf v; uint4v q[2]; };

__device__ __forceinline__ v8f vzero8() {
  v8f z = {0.f, 0.f, 0.f, 0.f, 0.f, 0.f, 0.f, 0.f};
  return z;
}

// fp32 -> bf16 bits, round-to-nearest-even
__device__ __forceinline__ unsigned short f2bf(float f) {
  unsigned int u = __float_as_uint(f);
  u = u + 0x7FFFu + ((u >> 16) & 1u);
  return (unsigned short)(u >> 16);
}

__device__ __forceinline__ float bf2f(unsigned short b) {
  return __uint_as_float(((unsigned int)b) << 16);
}

__device__ __forceinline__ float sigf(float x) {
  return 1.0f / (1.0f + __expf(-x));
}

__device__ __forceinline__ v8f wmma_bf16(const ABReg& a, const ABReg& b, v8f c) {
  return __builtin_amdgcn_wmma_f32_16x16x32_bf16(false, a.v, false, b.v,
                                                 (short)0, c, false, false);
}

// ---------------------------------------------------------------------------
// Elementwise fp32 -> bf16 conversion (grid-stride)
// ---------------------------------------------------------------------------
__global__ void cvt_f32_bf16(const float* __restrict__ src,
                             unsigned short* __restrict__ dst, int n) {
  int i = blockIdx.x * blockDim.x + threadIdx.x;
  int stride = gridDim.x * blockDim.x;
  for (; i < n; i += stride) dst[i] = f2bf(src[i]);
}

__global__ void zero_f32(float* __restrict__ p, int n) {
  int i = blockIdx.x * blockDim.x + threadIdx.x;
  if (i < n) p[i] = 0.0f;
}

// ---------------------------------------------------------------------------
// Kernel 1: input projection  XG[M,1024] = Xb[M,256] @ Wb[1024,256]^T + bsum
// Block = 256 thr (8 waves) -> 128(M) x 64(N) tile.
// The 64-row B band (32KB) is staged into LDS once per block with CDNA5
// async-to-LDS loads and shared by all 8 waves; A is fully preloaded into
// registers (64 VGPRs) so the K-loop is ds_load + wmma only.
// ---------------------------------------------------------------------------
__global__ void __launch_bounds__(256)
lstm_xproj(const unsigned short* __restrict__ Xb,   // [MROWS, 256] bf16
           const unsigned short* __restrict__ Wb,   // [1024, 256]  bf16
           const float* __restrict__ bx,            // [1024]
           const float* __restrict__ bh,            // [1024]
           const float* __restrict__ bb,            // [1024]
           float* __restrict__ XG)                  // [MROWS, 1024]
{
  __shared__ unsigned short Bs[64 * HPAD];          // ~33 KB, padded rows

  const int lane = threadIdx.x & 31;
  const int wave = threadIdx.x >> 5;
  const int hlf  = lane >> 4;     // K-half selector
  const int lr   = lane & 15;     // row (A) / col (B,D) within tile

  const int mtile = blockIdx.x * 8 + wave;   // 0..2047
  const int m0 = mtile * 16;
  const int n0 = blockIdx.y * 64;

  // ---- async-stage B band: rows n0..n0+63, full K, 2048 x 16B chunks ----
  {
    const unsigned short* gbase = Wb + (size_t)n0 * DDIM;
    #pragma unroll
    for (int it = 0; it < 8; ++it) {
      const int c   = it * 256 + threadIdx.x;   // chunk id
      const int row = c >> 5;                   // 32 chunks per 512B row
      const int ch  = c & 31;
      unsigned long long ga =
          (unsigned long long)(uintptr_t)(gbase + (size_t)row * DDIM + ch * 8);
      unsigned int la =
          (unsigned int)(uintptr_t)&Bs[row * HPAD + ch * 8];
      asm volatile("global_load_async_to_lds_b128 %0, %1, off"
                   :: "v"(la), "v"(ga) : "memory");
    }
    asm volatile("s_wait_asynccnt 0x0" ::: "memory");
  }

  // ---- preload all A k-chunks into registers (16 x b128) ----
  const unsigned short* arow = Xb + (size_t)(m0 + lr) * DDIM;
  ABReg A[8];
  #pragma unroll
  for (int k8 = 0; k8 < 8; ++k8) {
    A[k8].q[0] = *(const uint4v*)(arow + k8 * 32 + 8 * hlf);
    A[k8].q[1] = *(const uint4v*)(arow + k8 * 32 + 16 + 8 * hlf);
  }

  __syncthreads();   // B band visible to all waves

  v8f acc[4];
  #pragma unroll
  for (int nt = 0; nt < 4; ++nt) acc[nt] = vzero8();

  #pragma unroll
  for (int k8 = 0; k8 < 8; ++k8) {
    const int koff = k8 * 32 + 8 * hlf;
    #pragma unroll
    for (int nt = 0; nt < 4; ++nt) {
      const unsigned short* bls = Bs + (nt * 16 + lr) * HPAD + koff;
      ABReg Bm;
      Bm.q[0] = *(const uint4v*)(bls);
      Bm.q[1] = *(const uint4v*)(bls + 16);
      acc[nt] = wmma_bf16(A[k8], Bm, acc[nt]);
    }
  }

  // Epilogue: add fused biases, store D (m = r + 8*half, n = n0+nt*16+lr)
  #pragma unroll
  for (int nt = 0; nt < 4; ++nt) {
    const int n = n0 + nt * 16 + lr;
    const float bsum = bx[n] + bh[n] + bb[n];
    #pragma unroll
    for (int r = 0; r < 8; ++r) {
      const int m = m0 + r + 8 * hlf;
      XG[(size_t)m * G4 + n] = acc[nt][r] + bsum;
    }
  }
}

// ---------------------------------------------------------------------------
// Kernel 2: recurrence. One block per 16 batch rows (grid = 4, block = 512 =
// 16 waves). Wave w owns gate columns {0,256,512,768}+16w so each thread holds
// matching i,f,g,o cells in its 4 accumulators -> elementwise LSTM entirely in
// registers. h (bf16) goes through padded LDS between steps; c stays in
// registers; Wh stays hot in L2. B tiles are double-buffered one k-iteration
// ahead; next-step xg rows are prefetched during the WMMA chain.
// ---------------------------------------------------------------------------
__global__ void __launch_bounds__(512)
lstm_rec(const float* __restrict__ XG,              // [MROWS, 1024]
         const unsigned short* __restrict__ Whb,    // [1024, 256] bf16
         unsigned short* __restrict__ Ynext,        // [MROWS, 256] bf16
         float* __restrict__ Yout,                  // d_out [B,T,H] (if last)
         float* __restrict__ hcarry,                // [B,H]
         float* __restrict__ ccarry,                // [B,H]
         int last)
{
  __shared__ unsigned short hs[16 * HPAD];          // ~8.4 KB, padded rows

  const int tid  = threadIdx.x;
  const int lane = tid & 31;
  const int wave = tid >> 5;        // 0..15
  const int hlf  = lane >> 4;
  const int lr   = lane & 15;
  const int b0   = blockIdx.x * 16;
  const int j    = wave * 16 + lr;  // owned h/c column, 0..255

  // Load carried-in c into registers (thread owns cells (m = r+8*hlf, j))
  v8f c;
  #pragma unroll
  for (int r = 0; r < 8; ++r)
    c[r] = ccarry[(size_t)(b0 + r + 8 * hlf) * HDIM + j];

  // Load carried-in h into LDS as bf16 (padded rows)
  for (int i = tid; i < 16 * HDIM; i += 512) {
    const int m = i >> 8, col = i & 255;
    hs[m * HPAD + col] = f2bf(hcarry[(size_t)b0 * HDIM + i]);
  }
  __syncthreads();

  for (int t = 0; t < TLEN; ++t) {
    // Initialize the four gate accumulators from the precomputed projection
    v8f ai, af, ag, ao;
    #pragma unroll
    for (int r = 0; r < 8; ++r) {
      const float* xr = XG + ((size_t)(b0 + r + 8 * hlf) * TLEN + t) * G4;
      ai[r] = xr[j];
      af[r] = xr[j + 256];
      ag[r] = xr[j + 512];
      ao[r] = xr[j + 768];
    }

    // Prefetch next timestep's projection rows (overlaps the WMMA chain)
    if (t + 1 < TLEN) {
      #pragma unroll
      for (int r = 0; r < 8; ++r) {
        const float* xn = XG + ((size_t)(b0 + r + 8 * hlf) * TLEN + t + 1) * G4;
        __builtin_prefetch(xn + j,       0, 0);
        __builtin_prefetch(xn + j + 256, 0, 0);
        __builtin_prefetch(xn + j + 512, 0, 0);
        __builtin_prefetch(xn + j + 768, 0, 0);
      }
    }

    // Preload the full A (h) operand from LDS: 16 x ds_load_b128
    const unsigned short* hrow = hs + lr * HPAD;    // A row m = lr
    ABReg A[8];
    #pragma unroll
    for (int k8 = 0; k8 < 8; ++k8) {
      A[k8].q[0] = *(const uint4v*)(hrow + k8 * 32 + 8 * hlf);
      A[k8].q[1] = *(const uint4v*)(hrow + k8 * 32 + 16 + 8 * hlf);
    }

    // gates += h @ Wh^T ; B (Wh) double-buffered one k-iteration ahead
    ABReg Bc[4], Bn[4];
    {
      const int koff = 8 * hlf;
      #pragma unroll
      for (int g = 0; g < 4; ++g) {
        const unsigned short* w = Whb + (size_t)(j + g * 256) * HDIM + koff;
        Bc[g].q[0] = *(const uint4v*)(w);
        Bc[g].q[1] = *(const uint4v*)(w + 16);
      }
    }
    #pragma unroll
    for (int k8 = 0; k8 < 8; ++k8) {
      if (k8 < 7) {
        const int koff = (k8 + 1) * 32 + 8 * hlf;
        #pragma unroll
        for (int g = 0; g < 4; ++g) {
          const unsigned short* w = Whb + (size_t)(j + g * 256) * HDIM + koff;
          Bn[g].q[0] = *(const uint4v*)(w);
          Bn[g].q[1] = *(const uint4v*)(w + 16);
        }
      }
      ai = wmma_bf16(A[k8], Bc[0], ai);
      af = wmma_bf16(A[k8], Bc[1], af);
      ag = wmma_bf16(A[k8], Bc[2], ag);
      ao = wmma_bf16(A[k8], Bc[3], ao);
      #pragma unroll
      for (int g = 0; g < 4; ++g) Bc[g] = Bn[g];   // renamed away by unroll
    }

    // LSTM cell update, fully in registers
    v8f hnew;
    #pragma unroll
    for (int r = 0; r < 8; ++r) {
      const float iv = sigf(ai[r]);
      const float fv = sigf(af[r]);
      const float ov = sigf(ao[r]);
      const float gv = tanhf(ag[r]);
      const float cv = fv * c[r] + iv * gv;
      c[r] = cv;
      hnew[r] = ov * tanhf(cv);
    }

    __syncthreads();   // everyone finished reading hs for this step
    #pragma unroll
    for (int r = 0; r < 8; ++r) {
      const int m = r + 8 * hlf;
      const unsigned short hb = f2bf(hnew[r]);
      hs[m * HPAD + j] = hb;
      const size_t row = (size_t)(b0 + m) * TLEN + t;
      Ynext[row * HDIM + j] = hb;            // next layer's bf16 input
      if (last) Yout[row * HDIM + j] = hnew[r];
    }
    __syncthreads();   // new h visible before next step's A loads
  }

  // Persist carries (h,c flow into the next layer, per the reference)
  #pragma unroll
  for (int r = 0; r < 8; ++r)
    ccarry[(size_t)(b0 + r + 8 * hlf) * HDIM + j] = c[r];
  for (int i = tid; i < 16 * HDIM; i += 512) {
    const int m = i >> 8, col = i & 255;
    hcarry[(size_t)b0 * HDIM + i] = bf2f(hs[m * HPAD + col]);
  }
}

// ---------------------------------------------------------------------------
// Host-side orchestration
// ---------------------------------------------------------------------------
extern "C" void kernel_launch(void* const* d_in, const int* in_sizes, int n_in,
                              void* d_out, int out_size, void* d_ws, size_t ws_size,
                              hipStream_t stream) {
  (void)in_sizes; (void)n_in; (void)out_size; (void)ws_size;

  const float* x    = (const float*)d_in[0];   // [B,T,D]
  const float* Wx_w = (const float*)d_in[1];   // [3,1024,256]
  const float* Wx_b = (const float*)d_in[2];   // [3,1024]
  const float* Wh_w = (const float*)d_in[3];   // [3,1024,256]
  const float* Wh_b = (const float*)d_in[4];   // [3,1024]
  const float* bias = (const float*)d_in[5];   // [3,1024]

  // Workspace layout (~147 MB)
  char* ws = (char*)d_ws;
  float* xg = (float*)ws;                                         // 128 MB
  size_t off = (size_t)MROWS * G4 * sizeof(float);
  unsigned short* xb = (unsigned short*)(ws + off);               // 16 MB
  off += (size_t)MROWS * DDIM * sizeof(unsigned short);
  unsigned short* wxb = (unsigned short*)(ws + off);              // 1.5 MB
  off += (size_t)3 * G4 * DDIM * sizeof(unsigned short);
  unsigned short* whb = (unsigned short*)(ws + off);              // 1.5 MB
  off += (size_t)3 * G4 * HDIM * sizeof(unsigned short);
  float* hcarry = (float*)(ws + off);                             // 64 KB
  off += (size_t)BSZ * HDIM * sizeof(float);
  float* ccarry = (float*)(ws + off);                             // 64 KB

  // Zero h,c carries (hcarry & ccarry are adjacent: 2*B*H floats)
  zero_f32<<<(2 * BSZ * HDIM + 255) / 256, 256, 0, stream>>>(hcarry,
                                                             2 * BSZ * HDIM);

  // fp32 -> bf16 conversions (x once, weights once for all layers)
  cvt_f32_bf16<<<4096, 256, 0, stream>>>(x, xb, MROWS * DDIM);
  cvt_f32_bf16<<<1024, 256, 0, stream>>>(Wx_w, wxb, 3 * G4 * DDIM);
  cvt_f32_bf16<<<1024, 256, 0, stream>>>(Wh_w, whb, 3 * G4 * HDIM);

  for (int l = 0; l < 3; ++l) {
    lstm_xproj<<<dim3(256, 16), 256, 0, stream>>>(
        xb, wxb + (size_t)l * G4 * DDIM,
        Wx_b + (size_t)l * G4, Wh_b + (size_t)l * G4, bias + (size_t)l * G4,
        xg);
    lstm_rec<<<4, 512, 0, stream>>>(
        xg, whb + (size_t)l * G4 * HDIM,
        xb,                      // overwritten in place: becomes next layer input
        (float*)d_out, hcarry, ccarry, (l == 2) ? 1 : 0);
  }
}
